// ScaledDotProductAttention_180388627185
// MI455X (gfx1250) — matrix operand order repaired and checked by reading, not compiled, not run on hardware
//
#include <hip/hip_runtime.h>

// ---- types matching gfx1250 WMMA builtin signatures -------------------------
typedef _Float16 half_t;
typedef __attribute__((ext_vector_type(16))) _Float16 v16h;
typedef __attribute__((ext_vector_type(8)))  _Float16 v8h;
typedef __attribute__((ext_vector_type(4)))  _Float16 v4h;
typedef __attribute__((ext_vector_type(8)))  float    v8f;

// ---- problem constants (reference: B=2,H=16,S=2048,D=64) --------------------
constexpr int S_LEN   = 2048;
constexpr int D_HEAD  = 64;
constexpr int N_BH    = 2 * 16;
constexpr int BLOCK_M = 64;     // q rows per workgroup
constexpr int BLOCK_N = 64;     // keys per iteration
constexpr float QK_SCALE = 0.125f;  // 1/sqrt(64)

// A-fragment loader: 16x32 A in this file's contiguous k order (same as B).
// lane<16 holds K {kb..kb+15} with kb=0; lane>=16 with kb=16.
__device__ __forceinline__ v16h load_a_frag(const half_t* row, int kb) {
    v8h lo = *(const v8h*)(row + kb);
    v8h hi = *(const v8h*)(row + kb + 8);
    v16h a;
#pragma unroll
    for (int i = 0; i < 8; ++i) { a[i] = lo[i]; a[i + 8] = hi[i]; }
    return a;
}

__global__ __launch_bounds__(128)
void ScaledDotProductAttention_180388627185_kernel(
        const float* __restrict__ Q, const float* __restrict__ K,
        const float* __restrict__ V, const float* __restrict__ Mask,
        float* __restrict__ Out)
{
    __shared__ __align__(16) half_t Qs[BLOCK_M][D_HEAD];        // q rows (scaled, f16)
    __shared__ __align__(16) half_t Ks[BLOCK_N][D_HEAD];        // key rows
    __shared__ __align__(16) half_t Vt[D_HEAD][BLOCK_N];        // V transposed: Vt[d][k]
    __shared__ __align__(16) half_t Ps[4][16][BLOCK_N];         // per-wave prob tiles
    __shared__ __align__(16) float  Msk[BLOCK_M][BLOCK_N];      // mask block (async-staged fp32)

    const int tid    = threadIdx.x;
    const int w      = tid >> 5;        // wave id 0..3 -> 16-row stripe
    const int lane   = tid & 31;
    const int col    = lane & 15;       // N/col index in B/C/D layouts
    const int halfid = lane >> 4;       // lane half selects K range / row+8
    const int q0     = blockIdx.x * BLOCK_M;
    const int bh     = blockIdx.y;

    const float* Qb = Q + (size_t)bh * S_LEN * D_HEAD;
    const float* Kb = K + (size_t)bh * S_LEN * D_HEAD;
    const float* Vb = V + (size_t)bh * S_LEN * D_HEAD;
    float*       Ob = Out + (size_t)bh * S_LEN * D_HEAD;

    // LDS byte offset of the mask buffer (low 32 bits of the LDS-aperture addr)
    const unsigned msk_lds_base = (unsigned)(size_t)(void*)&Msk[0][0];
    // Uniform global base of this workgroup's mask rows (mask is [1,1,S,S])
    const unsigned long long msk_gbase =
        (unsigned long long)(const void*)(Mask + (size_t)q0 * S_LEN);

    // ---- stage Q tile (scale folded in) ------------------------------------
    {
        const float4* Qg4 = (const float4*)(Qb + (size_t)q0 * D_HEAD);
        for (int i = tid; i < BLOCK_M * D_HEAD / 4; i += 128) {
            float4 f = Qg4[i];
            int r = i >> 4, c = (i & 15) << 2;
            v4h h = { (half_t)(f.x * QK_SCALE), (half_t)(f.y * QK_SCALE),
                      (half_t)(f.z * QK_SCALE), (half_t)(f.w * QK_SCALE) };
            *(v4h*)&Qs[r][c] = h;
        }
    }

    // ---- per-lane running state (C/D layout: VGPR r -> row r + 8*halfid) ---
    v8f Oacc[4];
#pragma unroll
    for (int t = 0; t < 4; ++t) Oacc[t] = (v8f){};
    v8f m_run, l_run;
#pragma unroll
    for (int r = 0; r < 8; ++r) { m_run[r] = -1e30f; l_run[r] = 0.0f; }

    for (int kb0 = 0; kb0 < S_LEN; kb0 += BLOCK_N) {
        __syncthreads();   // previous iteration done reading Ks/Vt/Msk

        // ---- async-stage the fp32 mask block straight into LDS -------------
        // One wave-wide GLOBAL_LOAD_ASYNC_TO_LDS_B128 moves 512B; 8 per wave.
        for (int i = tid; i < BLOCK_M * BLOCK_N / 4; i += 128) {
            int r = i >> 4, c = (i & 15) << 2;
            unsigned voff   = (unsigned)(((size_t)r * S_LEN + kb0 + c) * 4);
            unsigned ldsoff = msk_lds_base + (unsigned)((r * BLOCK_N + c) * 4);
            asm volatile("global_load_async_to_lds_b128 %0, %1, %2"
                         :: "v"(ldsoff), "v"(voff), "s"(msk_gbase)
                         : "memory");
        }

        // ---- stage K block (row-major) and V block (transposed), f32->f16 --
        const float4* Kg4 = (const float4*)(Kb + (size_t)kb0 * D_HEAD);
        const float4* Vg4 = (const float4*)(Vb + (size_t)kb0 * D_HEAD);
        for (int i = tid; i < BLOCK_N * D_HEAD / 4; i += 128) {
            int r = i >> 4, c = (i & 15) << 2;
            float4 f = Kg4[i];
            v4h h = { (half_t)f.x, (half_t)f.y, (half_t)f.z, (half_t)f.w };
            *(v4h*)&Ks[r][c] = h;
            float4 g = Vg4[i];
            Vt[c + 0][r] = (half_t)g.x;
            Vt[c + 1][r] = (half_t)g.y;
            Vt[c + 2][r] = (half_t)g.z;
            Vt[c + 3][r] = (half_t)g.w;
        }
        if (kb0 + BLOCK_N < S_LEN) {
            // warm caches for next key block (global_prefetch_b8)
            __builtin_prefetch(Kb + (size_t)(kb0 + BLOCK_N) * D_HEAD + tid * 32, 0, 1);
            __builtin_prefetch(Vb + (size_t)(kb0 + BLOCK_N) * D_HEAD + tid * 32, 0, 1);
        }
        // all of this wave's async LDS writes must land before the barrier
        asm volatile("s_wait_asynccnt 0" ::: "memory");
        __syncthreads();

        // ---- scores: 4 tiles of 16x16, K-dim 64 = 2 WMMAs each -------------
        const half_t* arow_q = &Qs[w * 16 + col][0];
        v8f s[4];
#pragma unroll
        for (int nt = 0; nt < 4; ++nt) {
            v8f acc = (v8f){};
#pragma unroll
            for (int ks = 0; ks < 2; ++ks) {
                v16h a = load_a_frag(arow_q, ks * 32 + (halfid ? 16 : 0));
                // B layout: col = lane&15, contiguous 16 K vals per lane half
                v16h b = *(const v16h*)&Ks[nt * 16 + col][ks * 32 + halfid * 16];
                acc = __builtin_amdgcn_wmma_f32_16x16x32_f16(
                        false, a, false, b, (short)0, acc, false, false);
            }
            // add attention mask from LDS (C layout scatter -> ds_load_b32)
#pragma unroll
            for (int r = 0; r < 8; ++r)
                acc[r] += Msk[w * 16 + r + 8 * halfid][nt * 16 + col];
            s[nt] = acc;
        }

        // ---- online softmax (row reductions across 16-lane groups) ---------
        v8f mb;
#pragma unroll
        for (int r = 0; r < 8; ++r)
            mb[r] = fmaxf(fmaxf(s[0][r], s[1][r]), fmaxf(s[2][r], s[3][r]));
#pragma unroll
        for (int off = 8; off >= 1; off >>= 1)
#pragma unroll
            for (int r = 0; r < 8; ++r)
                mb[r] = fmaxf(mb[r], __shfl_xor(mb[r], off, 32));

        v8f mnew, alpha;
#pragma unroll
        for (int r = 0; r < 8; ++r) {
            mnew[r]  = fmaxf(m_run[r], mb[r]);
            alpha[r] = __expf(m_run[r] - mnew[r]);
        }

        v8f rowsum = (v8f){};
#pragma unroll
        for (int nt = 0; nt < 4; ++nt)
#pragma unroll
            for (int r = 0; r < 8; ++r) {
                float p = __expf(s[nt][r] - mnew[r]);
                rowsum[r] += p;
                Ps[w][r + 8 * halfid][nt * 16 + col] = (half_t)p;   // C->A relayout via LDS
            }
#pragma unroll
        for (int off = 8; off >= 1; off >>= 1)
#pragma unroll
            for (int r = 0; r < 8; ++r)
                rowsum[r] += __shfl_xor(rowsum[r], off, 32);

#pragma unroll
        for (int r = 0; r < 8; ++r) {
            l_run[r] = l_run[r] * alpha[r] + rowsum[r];
            m_run[r] = mnew[r];
        }
#pragma unroll
        for (int t = 0; t < 4; ++t)
#pragma unroll
            for (int r = 0; r < 8; ++r) Oacc[t][r] *= alpha[r];

        // ---- O += P x V : 2 K-slices x 4 d-tiles ---------------------------
        const half_t* arow_p = &Ps[w][col][0];
#pragma unroll
        for (int ks = 0; ks < 2; ++ks) {
            v16h a = load_a_frag(arow_p, ks * 32 + (halfid ? 16 : 0));
#pragma unroll
            for (int nt = 0; nt < 4; ++nt) {
                v16h b = *(const v16h*)&Vt[nt * 16 + col][ks * 32 + halfid * 16];
                Oacc[nt] = __builtin_amdgcn_wmma_f32_16x16x32_f16(
                        false, a, false, b, (short)0, Oacc[nt], false, false);
            }
        }
    }

    // ---- epilogue: divide by l, store fp32 ---------------------------------
#pragma unroll
    for (int r = 0; r < 8; ++r) {
        float inv = 1.0f / l_run[r];
        int q = q0 + w * 16 + r + 8 * halfid;
        float* orow = Ob + (size_t)q * D_HEAD;
        orow[ 0 + col] = Oacc[0][r] * inv;
        orow[16 + col] = Oacc[1][r] * inv;
        orow[32 + col] = Oacc[2][r] * inv;
        orow[48 + col] = Oacc[3][r] * inv;
    }
}

extern "C" void kernel_launch(void* const* d_in, const int* in_sizes, int n_in,
                              void* d_out, int out_size, void* d_ws, size_t ws_size,
                              hipStream_t stream) {
    const float* Q = (const float*)d_in[0];
    const float* K = (const float*)d_in[1];
    const float* V = (const float*)d_in[2];
    const float* M = (const float*)d_in[3];
    float* O = (float*)d_out;
    dim3 grid(S_LEN / BLOCK_M, N_BH);
    ScaledDotProductAttention_180388627185_kernel<<<grid, 128, 0, stream>>>(Q, K, V, M, O);
}
